// GraphConvNet_14070312862351
// MI455X (gfx1250) — matrix operand order repaired and checked by reading, not compiled, not run on hardware
//
#include <hip/hip_runtime.h>

typedef __attribute__((ext_vector_type(16))) _Float16 v16h;
typedef __attribute__((ext_vector_type(8)))  float    v8f;
typedef __attribute__((ext_vector_type(4)))  float    v4f;

#define H_DIM 96

// ---------------------------------------------------------------------------
// Pack a row-major f32 weight matrix B[K][Ncols] into per-lane f16 WMMA
// B-fragments.  Fragment layout (V_WMMA_F32_16X16X32_F16, wave32):
//   lane n      (0..15)  : column n of tile, K = kt*32 + 0..7, 16..23  (16 halves)
//   lane n+16            : column n of tile, K = kt*32 + 8..15, 24..31
// Blob layout: [(kt*NT + nt)*32 + lane] * 16 halves  (32B per lane chunk).
// ---------------------------------------------------------------------------
__global__ void pack_b_kernel(const float* __restrict__ B, _Float16* __restrict__ Bp,
                              int K, int Ncols)
{
    int idx = blockIdx.x * blockDim.x + threadIdx.x;
    int NT = Ncols >> 4;
    int KT = K >> 5;
    int total = KT * NT * 32;
    if (idx >= total) return;
    int lane = idx & 31;
    int tile = idx >> 5;
    int nt = tile % NT;
    int kt = tile / NT;
    int n  = nt * 16 + (lane & 15);
    int kb = kt * 32 + ((lane >> 4) << 3);
    _Float16* out = Bp + (size_t)idx * 16;
#pragma unroll
    for (int j = 0; j < 16; ++j)
        out[j] = (_Float16)B[(size_t)(kb + j + ((j >> 3) << 3)) * Ncols + n];
}

// Zero-padded copy (used to pad x so GEMM A-loads need no bounds checks).
__global__ void pad_copy_kernel(const float* __restrict__ src, float* __restrict__ dst,
                                int n_valid, int n_total)
{
    int idx = blockIdx.x * blockDim.x + threadIdx.x;
    if (idx >= n_total) return;
    dst[idx] = (idx < n_valid) ? src[idx] : 0.f;
}

// ---------------------------------------------------------------------------
// WMMA GEMM:  C[M x (NT*16)] = op(A) [M x K] * B [K x (NT*16)] + bias
//   FUSED=false: op(A) = A
//   FUSED=true : op(A) = (1+eps)*A + Aadd     (GINE combine fused in)
// All A/C buffers are padded to M % 64 == 0, so no bounds checks anywhere:
// straight float4 loads -> f16 fragment -> v_wmma.
// A fragment (16-bit A 16x32 layout): lane m (0..15): K = kb..kb+7, kb+16..kb+23
// with kb = kt*32;  lane m+16: same rows, kb = kt*32+8.
// ---------------------------------------------------------------------------
template <int NT, bool FUSED>
__global__ __launch_bounds__(128)
void gemm_wmma_kernel(const float* __restrict__ A, const float* __restrict__ Aadd,
                      const float* __restrict__ epsPtr,
                      const _Float16* __restrict__ Bp,
                      const float* __restrict__ bias,
                      float* __restrict__ C,
                      int K, int relu)
{
    const int lane = threadIdx.x & 31;
    const int wave = threadIdx.x >> 5;
    const int m0   = blockIdx.x * 64 + wave * 16;
    const int hi   = lane >> 4;
    const int row  = m0 + (lane & 15);
    const int Ncols = NT * 16;

    const float* arow   = A + (size_t)row * K;
    const float* addrow = FUSED ? (Aadd + (size_t)row * K) : nullptr;
    const float  alphav = FUSED ? (1.f + epsPtr[0]) : 1.f;

    v8f acc[NT];
    v8f zero = {};
#pragma unroll
    for (int t = 0; t < NT; ++t) acc[t] = zero;

    const int KT = K >> 5;
    for (int kt = 0; kt < KT; ++kt) {
        const int kb = kt * 32 + hi * 8;
        v4f a0 = *(const v4f*)(arow + kb);
        v4f a1 = *(const v4f*)(arow + kb + 4);
        v4f a2 = *(const v4f*)(arow + kb + 16);
        v4f a3 = *(const v4f*)(arow + kb + 20);
        if (FUSED) {
            const v4f b0 = *(const v4f*)(addrow + kb);
            const v4f b1 = *(const v4f*)(addrow + kb + 4);
            const v4f b2 = *(const v4f*)(addrow + kb + 16);
            const v4f b3 = *(const v4f*)(addrow + kb + 20);
            a0 = alphav * a0 + b0;
            a1 = alphav * a1 + b1;
            a2 = alphav * a2 + b2;
            a3 = alphav * a3 + b3;
        }
        v16h afrag;
        afrag[0]  = (_Float16)a0.x; afrag[1]  = (_Float16)a0.y;
        afrag[2]  = (_Float16)a0.z; afrag[3]  = (_Float16)a0.w;
        afrag[4]  = (_Float16)a1.x; afrag[5]  = (_Float16)a1.y;
        afrag[6]  = (_Float16)a1.z; afrag[7]  = (_Float16)a1.w;
        afrag[8]  = (_Float16)a2.x; afrag[9]  = (_Float16)a2.y;
        afrag[10] = (_Float16)a2.z; afrag[11] = (_Float16)a2.w;
        afrag[12] = (_Float16)a3.x; afrag[13] = (_Float16)a3.y;
        afrag[14] = (_Float16)a3.z; afrag[15] = (_Float16)a3.w;
#pragma unroll
        for (int t = 0; t < NT; ++t) {
            const v16h bfrag = *(const v16h*)(Bp + ((size_t)(kt * NT + t) * 32 + lane) * 16);
            acc[t] = __builtin_amdgcn_wmma_f32_16x16x32_f16(
                false, afrag, false, bfrag, (short)0, acc[t], false, false);
        }
    }

    // C/D layout: lane 0-15: N=lane, M=m0+vi; lane 16-31: N=lane-16, M=m0+8+vi
#pragma unroll
    for (int t = 0; t < NT; ++t) {
        const int n = t * 16 + (lane & 15);
        const float bv = bias ? bias[n] : 0.f;
#pragma unroll
        for (int vi = 0; vi < 8; ++vi) {
            const int m = m0 + hi * 8 + vi;
            float v = acc[t][vi] + bv;
            if (relu) v = fmaxf(v, 0.f);
            C[(size_t)m * Ncols + n] = v;
        }
    }
}

// ---------------------------------------------------------------------------
// Edge embedding:  e = relu(edge_attr[E,10] @ edge_W[10,96] + edge_b)
// Non-temporal stores: e is a 307MB stream consumed once per layer; keep the
// 192MB L2 for h/agg instead.
// ---------------------------------------------------------------------------
__global__ void edge_embed_kernel(const float* __restrict__ ea,
                                  const float* __restrict__ W,
                                  const float* __restrict__ b,
                                  float* __restrict__ e, int E)
{
    int idx = blockIdx.x * blockDim.x + threadIdx.x;
    if (idx >= E * 24) return;
    const int ed = idx / 24;
    const int f  = (idx % 24) * 4;
    float a[10];
#pragma unroll
    for (int k = 0; k < 10; ++k) a[k] = ea[(size_t)ed * 10 + k];
    float r0 = b[f], r1 = b[f + 1], r2 = b[f + 2], r3 = b[f + 3];
#pragma unroll
    for (int k = 0; k < 10; ++k) {
        const float* w = W + (size_t)k * H_DIM + f;
        r0 = fmaf(a[k], w[0], r0);
        r1 = fmaf(a[k], w[1], r1);
        r2 = fmaf(a[k], w[2], r2);
        r3 = fmaf(a[k], w[3], r3);
    }
    v4f o;
    o.x = fmaxf(r0, 0.f);
    o.y = fmaxf(r1, 0.f);
    o.z = fmaxf(r2, 0.f);
    o.w = fmaxf(r3, 0.f);
    __builtin_nontemporal_store(o, (v4f*)(e + (size_t)ed * H_DIM + f));
}

// ---------------------------------------------------------------------------
// GINE message + aggregate:  agg[dst] += relu(h[src] + e)
// h/agg (38MB each) are L2-resident on MI455X (192MB L2) -> gathers and
// fadd atomics stay in L2; e is streamed non-temporally from HBM.
// ---------------------------------------------------------------------------
__global__ void gather_scatter_kernel(const float* __restrict__ h,
                                      const float* __restrict__ ef,
                                      const int* __restrict__ src,
                                      const int* __restrict__ dst,
                                      float* __restrict__ agg, int E)
{
    int idx = blockIdx.x * blockDim.x + threadIdx.x;
    if (idx >= E * 24) return;
    const int ed = idx / 24;
    const int f  = (idx % 24) * 4;
    const int s = src[ed];
    const int d = dst[ed];
    const v4f hv = *(const v4f*)(h + (size_t)s * H_DIM + f);
    const v4f ev = __builtin_nontemporal_load(
        (const v4f*)(ef + (size_t)ed * H_DIM + f));
    float* ap = agg + (size_t)d * H_DIM + f;
    atomicAdd(ap + 0, fmaxf(hv.x + ev.x, 0.f));
    atomicAdd(ap + 1, fmaxf(hv.y + ev.y, 0.f));
    atomicAdd(ap + 2, fmaxf(hv.z + ev.z, 0.f));
    atomicAdd(ap + 3, fmaxf(hv.w + ev.w, 0.f));
}

// ---------------------------------------------------------------------------
// Per-graph node ranges (batch_idx is sorted).
// ---------------------------------------------------------------------------
__global__ void graph_range_init_kernel(int* gs, int* ge, int G)
{
    int i = blockIdx.x * blockDim.x + threadIdx.x;
    if (i < G) { gs[i] = 0; ge[i] = 0; }
}
__global__ void graph_range_kernel(const int* __restrict__ batch, int* gs, int* ge, int N)
{
    int i = blockIdx.x * blockDim.x + threadIdx.x;
    if (i >= N) return;
    const int b = batch[i];
    if (i == 0 || batch[i - 1] != b) gs[b] = i;
    if (i == N - 1 || batch[i + 1] != b) ge[b] = i + 1;
}

// ---------------------------------------------------------------------------
// GraphNorm stats: one workgroup per graph, one thread per feature.
// var(out) with out = z - mean*ms computed algebraically in a single pass:
//   var = E[z^2] - 2*a*E[z] + a^2,  a = mean*ms
// ---------------------------------------------------------------------------
__global__ void gn_stats_kernel(const float* __restrict__ z, const int* __restrict__ gs,
                                const int* __restrict__ ge, const float* __restrict__ ms,
                                float* __restrict__ ga, float* __restrict__ gi)
{
    const int g = blockIdx.x;
    const int f = threadIdx.x;
    const int s = gs[g], eN = ge[g];
    float sum = 0.f, sq = 0.f;
    for (int n = s; n < eN; ++n) {
        const float v = z[(size_t)n * H_DIM + f];
        sum += v;
        sq = fmaf(v, v, sq);
    }
    const float cnt  = fmaxf((float)(eN - s), 1.f);
    const float mean = sum / cnt;
    const float m2   = sq / cnt;
    const float a    = mean * ms[f];
    const float var  = m2 - 2.f * a * mean + a * a;
    ga[g * H_DIM + f] = a;
    gi[g * H_DIM + f] = rsqrtf(var + 1e-5f);
}

// GraphNorm apply + relu + residual (h updated in place).
__global__ void gn_apply_kernel(const float* __restrict__ z, const int* __restrict__ batch,
                                const float* __restrict__ ga, const float* __restrict__ gi,
                                const float* __restrict__ gw, const float* __restrict__ gb,
                                float* __restrict__ h, int N)
{
    int idx = blockIdx.x * blockDim.x + threadIdx.x;
    if (idx >= N * H_DIM) return;
    const int n = idx / H_DIM;
    const int f = idx % H_DIM;
    const int g = batch[n];
    const float o  = z[idx] - ga[g * H_DIM + f];
    const float zn = fmaf(gw[f] * o, gi[g * H_DIM + f], gb[f]);
    h[idx] = fmaxf(zn, 0.f) + h[idx];
}

// ---------------------------------------------------------------------------
// Pooling (mean | max) + per-row LayerNorm over 192 features.
// One workgroup (192 threads = 6 waves) per graph.
// ---------------------------------------------------------------------------
__global__ void pool_ln_kernel(const float* __restrict__ h, const int* __restrict__ gs,
                               const int* __restrict__ ge, const float* __restrict__ lng,
                               const float* __restrict__ lnb, float* __restrict__ gout)
{
    __shared__ float sg[2 * H_DIM];
    __shared__ float red[2];
    const int g = blockIdx.x;
    const int f = threadIdx.x;
    const int s = gs[g], eN = ge[g];
    float val;
    if (f < H_DIM) {
        float sum = 0.f;
        for (int n = s; n < eN; ++n) sum += h[(size_t)n * H_DIM + f];
        val = sum / fmaxf((float)(eN - s), 1.f);
    } else {
        const int ff = f - H_DIM;
        float mx = -3.402823466e38f;
        for (int n = s; n < eN; ++n) mx = fmaxf(mx, h[(size_t)n * H_DIM + ff]);
        val = mx;
    }
    sg[f] = val;
    __syncthreads();
    if (f == 0) {
        float s1 = 0.f;
        for (int i = 0; i < 2 * H_DIM; ++i) s1 += sg[i];
        const float mu = s1 / (2.f * H_DIM);
        float v = 0.f;
        for (int i = 0; i < 2 * H_DIM; ++i) { float d = sg[i] - mu; v += d * d; }
        red[0] = mu;
        red[1] = rsqrtf(v / (2.f * H_DIM) + 1e-5f);
    }
    __syncthreads();
    gout[(size_t)g * (2 * H_DIM) + f] = (val - red[0]) * red[1] * lng[f] + lnb[f];
}

// ---------------------------------------------------------------------------
extern "C" void kernel_launch(void* const* d_in, const int* in_sizes, int n_in,
                              void* d_out, int out_size, void* d_ws, size_t ws_size,
                              hipStream_t stream)
{
    const float* x         = (const float*)d_in[0];
    const float* edge_attr = (const float*)d_in[1];
    const int*   edge_idx  = (const int*)d_in[2];
    const int*   batch     = (const int*)d_in[3];
    const float* node_W    = (const float*)d_in[4];
    const float* node_b    = (const float*)d_in[5];
    const float* edge_W    = (const float*)d_in[6];
    const float* edge_b    = (const float*)d_in[7];
    const float* mlp_W1    = (const float*)d_in[8];
    const float* mlp_b1    = (const float*)d_in[9];
    const float* mlp_W2    = (const float*)d_in[10];
    const float* mlp_b2    = (const float*)d_in[11];
    const float* epsArr    = (const float*)d_in[12];
    const float* gn_w      = (const float*)d_in[13];
    const float* gn_b      = (const float*)d_in[14];
    const float* gn_ms     = (const float*)d_in[15];
    const float* ln_g      = (const float*)d_in[16];
    const float* ln_b      = (const float*)d_in[17];
    const float* out_W1    = (const float*)d_in[18];
    const float* out_b1    = (const float*)d_in[19];
    const float* out_W2    = (const float*)d_in[20];
    const float* out_b2    = (const float*)d_in[21];

    const int H   = H_DIM;
    const int N   = in_sizes[0] / 32;          // NI = 32
    const int E   = in_sizes[1] / 10;          // EI = 10
    const int L   = in_sizes[8] / (H * H);     // 3
    const int G   = out_size / 256;            // OUT = 256 -> G = 256
    const int Npad = ((N + 63) / 64) * 64;

    // ---- workspace carve-up --------------------------------------------
    char* ws = (char*)d_ws;
    size_t off = 0;
    auto alloc = [&](size_t bytes) -> char* {
        char* p = ws + off;
        off += (bytes + 255) & ~(size_t)255;
        return p;
    };
    float*    xpad   = (float*)alloc((size_t)Npad * 32 * 4);
    float*    hbuf   = (float*)alloc((size_t)Npad * H * 4);
    float*    agg    = (float*)alloc((size_t)Npad * H * 4);
    float*    tbuf   = (float*)alloc((size_t)Npad * H * 4);
    float*    z2     = (float*)alloc((size_t)Npad * H * 4);
    float*    ebuf   = (float*)alloc((size_t)E * H * 4);
    float*    galpha = (float*)alloc((size_t)G * H * 4);
    float*    ginv   = (float*)alloc((size_t)G * H * 4);
    float*    gbuf   = (float*)alloc((size_t)G * 2 * H * 4);
    float*    t2     = (float*)alloc((size_t)G * H * 4);
    int*      gs     = (int*)alloc((size_t)G * 4);
    int*      ge     = (int*)alloc((size_t)G * 4);
    _Float16* pbNode = (_Float16*)alloc((size_t)32 * H * 2);
    _Float16* pbW1   = (_Float16*)alloc((size_t)L * H * H * 2);
    _Float16* pbW2   = (_Float16*)alloc((size_t)L * H * H * 2);
    _Float16* pbO1   = (_Float16*)alloc((size_t)2 * H * H * 2);
    _Float16* pbO2   = (_Float16*)alloc((size_t)H * 256 * 2);
    (void)ws_size; (void)n_in;

    // ---- pack all weight matrices into WMMA B-fragment blobs -----------
    auto packLaunch = [&](const float* B, _Float16* Bp, int K, int Ncols) {
        int total = (K / 32) * (Ncols / 16) * 32;
        pack_b_kernel<<<dim3((total + 255) / 256), dim3(256), 0, stream>>>(B, Bp, K, Ncols);
    };
    packLaunch(node_W, pbNode, 32, H);
    for (int i = 0; i < L; ++i) {
        packLaunch(mlp_W1 + (size_t)i * H * H, pbW1 + (size_t)i * H * H, H, H);
        packLaunch(mlp_W2 + (size_t)i * H * H, pbW2 + (size_t)i * H * H, H, H);
    }
    packLaunch(out_W1, pbO1, 2 * H, H);
    packLaunch(out_W2, pbO2, H, 256);

    // ---- pad x so node-embed GEMM needs no bounds checks ----------------
    pad_copy_kernel<<<dim3((unsigned)((Npad * 32 + 255) / 256)), dim3(256), 0, stream>>>(
        x, xpad, N * 32, Npad * 32);

    // ---- per-graph node ranges ------------------------------------------
    graph_range_init_kernel<<<dim3((G + 255) / 256), dim3(256), 0, stream>>>(gs, ge, G);
    graph_range_kernel<<<dim3((N + 255) / 256), dim3(256), 0, stream>>>(batch, gs, ge, N);

    // ---- embeddings ------------------------------------------------------
    const dim3 gemmBlk(128);
    const dim3 gemmGridN(Npad / 64);
    gemm_wmma_kernel<6, false><<<gemmGridN, gemmBlk, 0, stream>>>(
        xpad, (const float*)nullptr, (const float*)nullptr, pbNode, node_b, hbuf, 32, 1);

    edge_embed_kernel<<<dim3((unsigned)((E * 24 + 255) / 256)), dim3(256), 0, stream>>>(
        edge_attr, edge_W, edge_b, ebuf, E);

    const int* src = edge_idx;
    const int* dst = edge_idx + E;

    // ---- L GINE layers ---------------------------------------------------
    for (int i = 0; i < L; ++i) {
        (void)hipMemsetAsync(agg, 0, (size_t)Npad * H * 4, stream);
        gather_scatter_kernel<<<dim3((unsigned)((E * 24 + 255) / 256)), dim3(256), 0, stream>>>(
            hbuf, ebuf, src, dst, agg, E);
        // t = relu(((1+eps)*h + agg) @ W1 + b1)   (z fused into A-load)
        gemm_wmma_kernel<6, true><<<gemmGridN, gemmBlk, 0, stream>>>(
            hbuf, agg, epsArr + i, pbW1 + (size_t)i * H * H, mlp_b1 + i * H, tbuf, H, 1);
        // z2 = t @ W2 + b2
        gemm_wmma_kernel<6, false><<<gemmGridN, gemmBlk, 0, stream>>>(
            tbuf, (const float*)nullptr, (const float*)nullptr,
            pbW2 + (size_t)i * H * H, mlp_b2 + i * H, z2, H, 0);
        gn_stats_kernel<<<dim3(G), dim3(H), 0, stream>>>(z2, gs, ge, gn_ms + i * H, galpha, ginv);
        gn_apply_kernel<<<dim3((unsigned)(((size_t)N * H + 255) / 256)), dim3(256), 0, stream>>>(
            z2, batch, galpha, ginv, gn_w + i * H, gn_b + i * H, hbuf, N);
    }

    // ---- readout ---------------------------------------------------------
    pool_ln_kernel<<<dim3(G), dim3(2 * H), 0, stream>>>(hbuf, gs, ge, ln_g, ln_b, gbuf);
    gemm_wmma_kernel<6, false><<<dim3(G / 64), gemmBlk, 0, stream>>>(
        gbuf, (const float*)nullptr, (const float*)nullptr, pbO1, out_b1, t2, 2 * H, 1);
    gemm_wmma_kernel<16, false><<<dim3(G / 64), gemmBlk, 0, stream>>>(
        t2, (const float*)nullptr, (const float*)nullptr, pbO2, out_b2, (float*)d_out, H, 0);
}
